// CIC_69861938037039
// MI455X (gfx1250) — matrix operand order repaired and checked by reading, not compiled, not run on hardware
//
#include <hip/hip_runtime.h>
#include <math.h>

// ---------------- problem constants ----------------
#define Bn    8
#define Nn    2048
#define NP    1024
#define KK    32
#define CIN   128
#define Pp    64
#define MIDc  32
#define COUT  256
#define CN    100
#define CL    5
#define RAD2  0.04f
#define EPSf  1e-5f

typedef __attribute__((ext_vector_type(16))) _Float16 v16h;
typedef __attribute__((ext_vector_type(8)))  float    v8f;

// =====================================================================
// Generic batched GEMM:  C[bz] = A[bz](MxK) * B[bz](KxN), fp32 in/out,
// f16 staged in LDS, V_WMMA_F32_16X16X32_F16 compute.
// A element (m,k) at A + m*lda + k   (transA=0)
//                 at A + k*lda + m   (transA=1)
// B element (k,n) at B + k*ldb + n.  C row-major ldc.
// Uniform fast-path staging when the tile is fully in bounds.
// =====================================================================
#define BM 32
#define BN 64
#define BK 32

__global__ void gemm_wmma_k(const float* __restrict__ A, const float* __restrict__ B,
                            float* __restrict__ C, int M, int N, int K,
                            int lda, int ldb, int ldc,
                            long long sA, long long sB, long long sC, int transA)
{
  const float* Ab = A + (long long)blockIdx.z * sA;
  const float* Bb = B + (long long)blockIdx.z * sB;
  float*       Cb = C + (long long)blockIdx.z * sC;
  int m0 = blockIdx.y * BM, n0 = blockIdx.x * BN;

  __shared__ alignas(16) _Float16 As[BM][BK];   // [m][k]
  __shared__ alignas(16) _Float16 Bs[BN][BK];   // transposed: [n][k]

  int tid  = threadIdx.x;           // 128 threads = 4 waves
  int lane = tid & 31, wave = tid >> 5;
  int wm = wave & 1;                // M tile 0..1
  int wn = (wave >> 1) << 1;        // N tile base 0 or 2
  int hi = lane >> 4, lo = lane & 15;

  const bool fullM = (m0 + BM <= M);
  const bool fullN = (n0 + BN <= N);

  v8f acc0 = {}; v8f acc1 = {};

  for (int k0 = 0; k0 < K; k0 += BK) {
    const bool fullK = (k0 + BK <= K);

    // ---------------- stage A (zero-padded) ----------------
    if (fullM && fullK && !transA) {
      // branch-free: one 8-wide contiguous K chunk per thread -> ds_store_b128
      int r = tid >> 2, c = (tid & 3) * 8;
      const float* src = Ab + (long long)(m0 + r) * lda + (k0 + c);
      union { _Float16 h[8]; float4 f; } pa;
#pragma unroll
      for (int j = 0; j < 8; ++j) pa.h[j] = (_Float16)src[j];
      *(float4*)&As[r][c] = pa.f;
    } else {
      for (int i = tid; i < BM * BK; i += 128) {
        int r = i >> 5, c = i & 31;
        int gm = m0 + r, gk = k0 + c;
        float v = 0.f;
        if (gm < M && gk < K)
          v = transA ? Ab[(long long)gk * lda + gm] : Ab[(long long)gm * lda + gk];
        As[r][c] = (_Float16)v;
      }
    }

    // ---------------- stage B transposed (zero-padded) ----------------
    if (fullN && fullK) {
      // branch-free: 4 chunks of 4 consecutive n at fixed k per thread
#pragma unroll
      for (int it = 0; it < 4; ++it) {
        int chunk = tid + it * 128;       // 0..511
        int kk = chunk >> 4;              // 0..31
        int nb = (chunk & 15) * 4;        // 0,4,...,60
        const float* src = Bb + (long long)(k0 + kk) * ldb + (n0 + nb);
#pragma unroll
        for (int t = 0; t < 4; ++t) Bs[nb + t][kk] = (_Float16)src[t];
      }
      if (k0 + BK < K)
        __builtin_prefetch(Bb + (long long)(k0 + BK + (tid >> 2)) * ldb + n0 + (tid & 3) * 16, 0, 1);
    } else {
      for (int i = tid; i < BN * BK; i += 128) {
        int n = i >> 5, c = i & 31;
        int gn = n0 + n, gk = k0 + c;
        float v = 0.f;
        if (gn < N && gk < K) v = Bb[(long long)gk * ldb + gn];
        Bs[n][c] = (_Float16)v;
      }
    }
    __syncthreads();

    union { v16h v; float4 f[2]; } af, bf0, bf1;
    // A frag: row = 16*wm + lo, K = 16*hi + j (contiguous 16)
    af.f[0] = *(const float4*)&As[16 * wm + lo][16 * hi];
    af.f[1] = *(const float4*)&As[16 * wm + lo][16 * hi + 8];
    // B frags: col = lo, K = 16*hi + j (contiguous 16)
    bf0.f[0] = *(const float4*)&Bs[16 * wn + lo][16 * hi];
    bf0.f[1] = *(const float4*)&Bs[16 * wn + lo][16 * hi + 8];
    bf1.f[0] = *(const float4*)&Bs[16 * (wn + 1) + lo][16 * hi];
    bf1.f[1] = *(const float4*)&Bs[16 * (wn + 1) + lo][16 * hi + 8];

    acc0 = __builtin_amdgcn_wmma_f32_16x16x32_f16(false, af.v, false, bf0.v,
                                                  (short)0, acc0, false, false);
    acc1 = __builtin_amdgcn_wmma_f32_16x16x32_f16(false, af.v, false, bf1.v,
                                                  (short)0, acc1, false, false);
    __syncthreads();
  }

#pragma unroll
  for (int r = 0; r < 8; ++r) {
    int gm = m0 + 16 * wm + r + 8 * hi;
    if (gm < M) {
      int gn0 = n0 + 16 * wn + lo;
      int gn1 = n0 + 16 * (wn + 1) + lo;
      if (gn0 < N) Cb[(long long)gm * ldc + gn0] = acc0[r];
      if (gn1 < N) Cb[(long long)gm * ldc + gn1] = acc1[r];
    }
  }
}

// =====================================================================
// BN helpers (deterministic reductions; stats = [mean,rstd] per channel)
// =====================================================================
__global__ void bn_stats_k(const float* __restrict__ x, float* __restrict__ stats,
                           int Bc, int C, long long S)
{
  int c = blockIdx.x;
  __shared__ float s1s[256], s2s[256];
  float s1 = 0.f, s2 = 0.f;
  long long tot = (long long)Bc * S;
  for (long long i = threadIdx.x; i < tot; i += blockDim.x) {
    long long b = i / S, s = i % S;
    float v = x[(b * C + c) * S + s];
    s1 += v; s2 += v * v;
  }
  s1s[threadIdx.x] = s1; s2s[threadIdx.x] = s2;
  __syncthreads();
  for (int off = 128; off > 0; off >>= 1) {
    if (threadIdx.x < off) { s1s[threadIdx.x] += s1s[threadIdx.x + off];
                             s2s[threadIdx.x] += s2s[threadIdx.x + off]; }
    __syncthreads();
  }
  if (threadIdx.x == 0) {
    float inv = 1.f / (float)tot;
    float m = s1s[0] * inv;
    float var = s2s[0] * inv - m * m;
    stats[2 * c]     = m;
    stats[2 * c + 1] = rsqrtf(var + EPSf);
  }
}

__global__ void bn_apply_k(const float* __restrict__ x, const float* __restrict__ res,
                           float* __restrict__ out, const float* __restrict__ stats,
                           const float* __restrict__ g, const float* __restrict__ bta,
                           int C, long long S, long long total, int do_lrelu)
{
  long long i = (long long)blockIdx.x * blockDim.x + threadIdx.x;
  if (i >= total) return;
  int c = (int)((i / S) % C);
  float v = (x[i] - stats[2 * c]) * stats[2 * c + 1] * g[c] + bta[c];
  if (res) v += res[i];
  if (do_lrelu) v = v > 0.f ? v : 0.2f * v;
  out[i] = v;
}

// =====================================================================
// FPS (sequential scan, one block per batch, matches jax.lax.scan emit order)
// =====================================================================
__global__ void fps_k(const float* __restrict__ xyz, int* __restrict__ fidx)
{
  int b = blockIdx.x, t = threadIdx.x;       // 1024 threads
  const float* X = xyz + (long long)b * 3 * Nn;
  __shared__ float dist[Nn];
  __shared__ float rv[1024];
  __shared__ int   ri[1024];
  __shared__ int   s_far;
  dist[t] = 1e10f; dist[t + 1024] = 1e10f;
  if (t == 0) s_far = 0;
  __syncthreads();
  for (int it = 0; it < NP; ++it) {
    int far = s_far;
    if (t == 0) fidx[b * NP + it] = far;
    float cx = X[far], cy = X[Nn + far], cz = X[2 * Nn + far];
    float best = -1.f; int bi = 0;
    for (int n = t; n < Nn; n += 1024) {
      float dx = X[n] - cx, dy = X[Nn + n] - cy, dz = X[2 * Nn + n] - cz;
      float d = fminf(dist[n], dx * dx + dy * dy + dz * dz);
      dist[n] = d;
      if (d > best) { best = d; bi = n; }
    }
    rv[t] = best; ri[t] = bi;
    __syncthreads();
    for (int off = 512; off > 0; off >>= 1) {
      if (t < off) {
        if (rv[t + off] > rv[t] || (rv[t + off] == rv[t] && ri[t + off] < ri[t])) {
          rv[t] = rv[t + off]; ri[t] = ri[t + off];
        }
      }
      __syncthreads();
    }
    if (t == 0) s_far = ri[0];
    __syncthreads();
  }
}

// new_xyz (B,NP,3) row-major + xyz_d (B,3,NP) into d_out head
__global__ void newxyz_k(const float* __restrict__ xyz, const int* __restrict__ fidx,
                         float* __restrict__ nxyz, float* __restrict__ out_xyzd)
{
  int i = blockIdx.x * blockDim.x + threadIdx.x;
  if (i >= Bn * NP) return;
  int b = i / NP, s = i % NP, j = fidx[i];
  for (int c = 0; c < 3; ++c) {
    float v = xyz[((long long)b * 3 + c) * Nn + j];
    nxyz[(long long)i * 3 + c] = v;
    out_xyzd[((long long)b * 3 + c) * NP + s] = v;
  }
}

// ball query: one wave per query, in-index-order first-K within radius
__global__ void ballq_k(const float* __restrict__ xyz, const float* __restrict__ nxyz,
                        int* __restrict__ gidx)
{
  int lane = threadIdx.x & 31;
  int q = blockIdx.x * (blockDim.x >> 5) + (threadIdx.x >> 5);
  if (q >= Bn * NP) return;
  int b = q / NP;
  const float* X = xyz + (long long)b * 3 * Nn;
  float qx = nxyz[(long long)q * 3], qy = nxyz[(long long)q * 3 + 1], qz = nxyz[(long long)q * 3 + 2];
  int* out = gidx + (long long)q * KK;
  int cnt = 0, firstn = 0;
  for (int base = 0; base < Nn && cnt < KK; base += 32) {
    int n = base + lane;
    float dx = X[n] - qx, dy = X[Nn + n] - qy, dz = X[2 * Nn + n] - qz;
    bool ok = (dx * dx + dy * dy + dz * dz) <= RAD2;
    unsigned mask = (unsigned)__ballot(ok);
    if (cnt == 0 && mask != 0u) firstn = base + __ffs(mask) - 1;
    int pos = cnt + __popc(mask & ((1u << lane) - 1u));
    if (ok && pos < KK) out[pos] = n;
    cnt += __popc(mask);
  }
  if (cnt > KK) cnt = KK;
  for (int j = cnt + lane; j < KK; j += 32) out[j] = firstn;
}

// grouped max-pool: block per (b,s), 128 threads over channels
__global__ void groupmax_k(const float* __restrict__ x, const int* __restrict__ gidx,
                           float* __restrict__ xn)
{
  int q = blockIdx.x, b = q / NP, s = q % NP, c = threadIdx.x;
  __shared__ int sidx[KK];
  if (c < KK) sidx[c] = gidx[(long long)q * KK + c];
  __syncthreads();
  const float* xb = x + ((long long)b * CIN + c) * Nn;
  float m = -3.4e38f;
  for (int k = 0; k < KK; ++k) m = fmaxf(m, xb[sidx[k]]);
  xn[((long long)b * CIN + c) * NP + s] = m;
}

// =====================================================================
// misc elementwise
// =====================================================================
__global__ void xatt_k(const float* __restrict__ h, const float* __restrict__ w,
                       float* __restrict__ xatt)
{
  int i = blockIdx.x * blockDim.x + threadIdx.x;
  if (i >= Bn * NP) return;
  int b = i / NP, n = i % NP;
  float v = 0.f;
  for (int c = 0; c < Pp; ++c) v += w[c] * h[((long long)b * Pp + c) * NP + n];
  xatt[i] = 1.f / (1.f + expf(-v));
}

__global__ void scale_hx_k(const float* __restrict__ h, const float* __restrict__ xatt,
                           float* __restrict__ hx)
{
  long long i = (long long)blockIdx.x * blockDim.x + threadIdx.x;
  if (i >= (long long)Bn * Pp * NP) return;
  int n = (int)(i % NP); int b = (int)(i / ((long long)Pp * NP));
  hx[i] = h[i] * xatt[b * NP + n];
}

// kNN (33 nearest incl self) on sampled points: block per query
__global__ void knn_k(const float* __restrict__ nxyz, int* __restrict__ idx33)
{
  int q = blockIdx.x, b = q / NP, s = q % NP;
  __shared__ float neg[NP];
  __shared__ float rv[256];
  __shared__ int   ri[256];
  const float* Pb = nxyz + (long long)b * NP * 3;
  float qx = Pb[s * 3], qy = Pb[s * 3 + 1], qz = Pb[s * 3 + 2];
  for (int n = threadIdx.x; n < NP; n += 256) {
    float dx = Pb[n * 3] - qx, dy = Pb[n * 3 + 1] - qy, dz = Pb[n * 3 + 2] - qz;
    neg[n] = -(dx * dx + dy * dy + dz * dz);
  }
  __syncthreads();
  for (int j = 0; j < KK + 1; ++j) {
    float best = -3.4e38f; int bi = 0;
    for (int n = threadIdx.x; n < NP; n += 256)
      if (neg[n] > best || (neg[n] == best && n < bi)) { best = neg[n]; bi = n; }
    rv[threadIdx.x] = best; ri[threadIdx.x] = bi;
    __syncthreads();
    for (int off = 128; off > 0; off >>= 1) {
      if (threadIdx.x < off) {
        if (rv[threadIdx.x + off] > rv[threadIdx.x] ||
            (rv[threadIdx.x + off] == rv[threadIdx.x] && ri[threadIdx.x + off] < ri[threadIdx.x])) {
          rv[threadIdx.x] = rv[threadIdx.x + off]; ri[threadIdx.x] = ri[threadIdx.x + off];
        }
      }
      __syncthreads();
    }
    if (threadIdx.x == 0) { idx33[(long long)q * 33 + j] = ri[0]; neg[ri[0]] = -3.4e38f; }
    __syncthreads();
  }
}

// top-CN start points per batch (descending, index tie-break like lax.top_k)
__global__ void topk_start_k(const float* __restrict__ xatt, int* __restrict__ start)
{
  int b = blockIdx.x;
  __shared__ float vals[NP];
  __shared__ float rv[256];
  __shared__ int   ri[256];
  for (int n = threadIdx.x; n < NP; n += 256) vals[n] = xatt[b * NP + n];
  __syncthreads();
  for (int j = 0; j < CN; ++j) {
    float best = -3.4e38f; int bi = 0;
    for (int n = threadIdx.x; n < NP; n += 256)
      if (vals[n] > best || (vals[n] == best && n < bi)) { best = vals[n]; bi = n; }
    rv[threadIdx.x] = best; ri[threadIdx.x] = bi;
    __syncthreads();
    for (int off = 128; off > 0; off >>= 1) {
      if (threadIdx.x < off) {
        if (rv[threadIdx.x + off] > rv[threadIdx.x] ||
            (rv[threadIdx.x + off] == rv[threadIdx.x] && ri[threadIdx.x + off] < ri[threadIdx.x])) {
          rv[threadIdx.x] = rv[threadIdx.x + off]; ri[threadIdx.x] = ri[threadIdx.x + off];
        }
      }
      __syncthreads();
    }
    if (threadIdx.x == 0) { start[b * CN + j] = ri[0]; vals[ri[0]] = -3.4e38f; }
    __syncthreads();
  }
}

// =====================================================================
// curve walk
// =====================================================================
__global__ void walk_init_k(const float* __restrict__ hx, const int* __restrict__ start,
                            float* __restrict__ pre, int* __restrict__ cur_idx)
{
  int i = blockIdx.x * blockDim.x + threadIdx.x;
  if (i >= Bn * Pp * CN) return;
  int s = i % CN, c = (i / CN) % Pp, b = i / (Pp * CN);
  int j = start[b * CN + s];
  pre[i] = hx[((long long)b * Pp + c) * NP + j];
  if (c == 0) cur_idx[b * CN + s] = j;
}

__global__ void walk_mom1_k(const float* __restrict__ curP, const float* __restrict__ pre,
                            const float* __restrict__ mw, float* __restrict__ mo)
{
  int i = blockIdx.x * blockDim.x + threadIdx.x;
  if (i >= Bn * CN) return;
  int b = i / CN, s = i % CN;
  float m0 = 0.f, m1 = 0.f;
  for (int c = 0; c < Pp; ++c) {
    float cf = curP[((long long)b * Pp + c) * CN + s];
    float pr = pre [((long long)b * Pp + c) * CN + s];
    m0 += mw[c] * cf + mw[Pp + c] * pr;
    m1 += mw[2 * Pp + c] * cf + mw[2 * Pp + Pp + c] * pr;
  }
  mo[((long long)b * 2 + 0) * CN + s] = m0;
  mo[((long long)b * 2 + 1) * CN + s] = m1;
}

__global__ void walk_mom2_k(const float* __restrict__ mo, const float* __restrict__ st,
                            const float* __restrict__ mg, const float* __restrict__ mb,
                            const float* __restrict__ curP, float* __restrict__ pre)
{
  int i = blockIdx.x * blockDim.x + threadIdx.x;
  if (i >= Bn * Pp * CN) return;
  int s = i % CN, b = i / (Pp * CN);
  float a0 = (mo[((long long)b * 2 + 0) * CN + s] - st[0]) * st[1] * mg[0] + mb[0];
  float a1 = (mo[((long long)b * 2 + 1) * CN + s] - st[2]) * st[3] * mg[1] + mb[1];
  float mx = fmaxf(a0, a1);
  float e0 = expf(a0 - mx), e1 = expf(a1 - mx), inv = 1.f / (e0 + e1);
  pre[i] = (e0 * inv) * curP[i] + (e1 * inv) * pre[i];
}

// raw agent logits (B,CN,KK)
__global__ void walk_agent1_k(const float* __restrict__ hx, const float* __restrict__ pre,
                              const int* __restrict__ idx33, const int* __restrict__ cur_idx,
                              const float* __restrict__ aw, float* __restrict__ logits)
{
  int q = blockIdx.x, b = q / CN, s = q % CN, k = threadIdx.x; // 32 threads
  int cprev = cur_idx[b * CN + s];
  int nb = idx33[((long long)b * NP + cprev) * 33 + 1 + k];
  float l = 0.f;
  for (int c = 0; c < Pp; ++c)
    l += aw[c] * hx[((long long)b * Pp + c) * NP + nb]
       + aw[Pp + c] * pre[((long long)b * Pp + c) * CN + s];
  logits[(long long)q * KK + k] = l;
}

// bn + optional direction weight + hard argmax select + feature update
__global__ void walk_agent2_k(const float* __restrict__ hx, const float* __restrict__ pre,
                              const int* __restrict__ idx33, int* __restrict__ cur_idx,
                              const float* __restrict__ logits, const float* __restrict__ st,
                              const float* __restrict__ ag, const float* __restrict__ ab,
                              const float* __restrict__ curP, float* __restrict__ curN,
                              float* __restrict__ curves, int step)
{
  int q = blockIdx.x, b = q / CN, s = q % CN, k = threadIdx.x;
  __shared__ float lv[KK];
  __shared__ int   lk[KK];
  __shared__ int   s_new;
  int cprev = cur_idx[b * CN + s];
  int nbk = idx33[((long long)b * NP + cprev) * 33 + 1 + k];
  float l = (logits[(long long)q * KK + k] - st[0]) * st[1] * ag[0] + ab[0];
  if (step > 0) {
    float dot = 0.f, na2 = 0.f, nb2 = 0.f;
    for (int c = 0; c < Pp; ++c) {
      float cv = curP[((long long)b * Pp + c) * CN + s];
      float av = cv - pre[((long long)b * Pp + c) * CN + s];
      float nv = hx[((long long)b * Pp + c) * NP + nbk] - cv;
      dot += av * nv; na2 += av * av; nb2 += nv * nv;
    }
    float den = fmaxf(sqrtf(na2) * sqrtf(nb2), 1e-8f);
    float d = 1.f + dot / den;
    l *= fminf(fmaxf(d, 0.f), 1.f);
  }
  lv[k] = l; lk[k] = k;
  __syncthreads();
  for (int off = 16; off > 0; off >>= 1) {
    if (k < off) {
      if (lv[k + off] > lv[k]) { lv[k] = lv[k + off]; lk[k] = lk[k + off]; }
    }
    __syncthreads();
  }
  if (k == 0) {
    int kstar = lk[0];
    int nc = idx33[((long long)b * NP + cprev) * 33 + 1 + kstar];
    cur_idx[b * CN + s] = nc;
    s_new = nc;
  }
  __syncthreads();
  int nc = s_new;
  for (int c = k; c < Pp; c += KK) {
    float v = hx[((long long)b * Pp + c) * NP + nc];
    curN[((long long)b * Pp + c) * CN + s] = v;
    curves[(((long long)b * Pp + c) * CN + s) * CL + step] = v;
  }
}

// =====================================================================
// curve aggregation small kernels
// =====================================================================
__global__ void ca_att_k(const float* __restrict__ curves, const float* __restrict__ la,
                         float* __restrict__ att)
{
  int i = blockIdx.x * blockDim.x + threadIdx.x;
  if (i >= Bn * CN * CL) return;
  int l = i % CL, s = (i / CL) % CN, b = i / (CL * CN);
  float v = 0.f;
  for (int c = 0; c < Pp; ++c) v += la[c] * curves[(((long long)b * Pp + c) * CN + s) * CL + l];
  att[i] = v;
}

__global__ void ca_sml_k(const float* __restrict__ att, float* __restrict__ sml)
{
  int i = blockIdx.x * blockDim.x + threadIdx.x;
  if (i >= Bn * CN) return;
  const float* p = att + (long long)i * CL;
  float mx = -3.4e38f;
  for (int l = 0; l < CL; ++l) mx = fmaxf(mx, p[l]);
  float s = 0.f;
  for (int l = 0; l < CL; ++l) s += expf(p[l] - mx);
  for (int l = 0; l < CL; ++l) sml[(long long)i * CL + l] = expf(p[l] - mx) / s;
}

__global__ void ca_sms_k(const float* __restrict__ att, float* __restrict__ sms)
{
  int i = blockIdx.x * blockDim.x + threadIdx.x;
  if (i >= Bn * CL) return;
  int b = i / CL, l = i % CL;
  const float* p = att + (long long)b * CN * CL + l;
  float mx = -3.4e38f;
  for (int s = 0; s < CN; ++s) mx = fmaxf(mx, p[s * CL]);
  float sum = 0.f;
  for (int s = 0; s < CN; ++s) sum += expf(p[s * CL] - mx);
  for (int s = 0; s < CN; ++s)
    sms[((long long)b * CN + s) * CL + l] = expf(p[s * CL] - mx) / sum;
}

__global__ void ca_inter0_k(const float* __restrict__ curves, const float* __restrict__ sml,
                            float* __restrict__ inter0)
{
  int i = blockIdx.x * blockDim.x + threadIdx.x;
  if (i >= Bn * Pp * CN) return;
  int s = i % CN, c = (i / CN) % Pp, b = i / (Pp * CN);
  float v = 0.f;
  for (int l = 0; l < CL; ++l)
    v += curves[(((long long)b * Pp + c) * CN + s) * CL + l] * sml[((long long)b * CN + s) * CL + l];
  inter0[i] = v;
}

__global__ void ca_intra0_k(const float* __restrict__ curves, const float* __restrict__ sms,
                            float* __restrict__ intra0)
{
  int i = blockIdx.x * blockDim.x + threadIdx.x;
  if (i >= Bn * Pp * CL) return;
  int l = i % CL, c = (i / CL) % Pp, b = i / (Pp * CL);
  float v = 0.f;
  for (int s = 0; s < CN; ++s)
    v += curves[(((long long)b * Pp + c) * CN + s) * CL + l] * sms[((long long)b * CN + s) * CL + l];
  intra0[i] = v;
}

// column-wise softmax over M rows (stride ldx) per (batch, col)
__global__ void softmax_col_k(float* __restrict__ X, int Mr, int ldx,
                              long long per_batch, int ncols, int batch)
{
  int i = blockIdx.x * blockDim.x + threadIdx.x;
  if (i >= batch * ncols) return;
  float* p = X + (long long)(i / ncols) * per_batch + (i % ncols);
  float mx = -3.4e38f;
  for (int j = 0; j < Mr; ++j) mx = fmaxf(mx, p[(long long)j * ldx]);
  float s = 0.f;
  for (int j = 0; j < Mr; ++j) s += expf(p[(long long)j * ldx] - mx);
  float inv = 1.f / s;
  for (int j = 0; j < Mr; ++j) p[(long long)j * ldx] = expf(p[(long long)j * ldx] - mx) * inv;
}

// =====================================================================
// LPFA
// =====================================================================
__global__ void pf_build_k(const float* __restrict__ nxyz, const int* __restrict__ idx33,
                           float* __restrict__ pf)
{
  int i = blockIdx.x * blockDim.x + threadIdx.x;
  if (i >= Bn * NP * KK) return;
  int k = i % KK, n = (i / KK) % NP, b = i / (KK * NP);
  const float* pz = nxyz + ((long long)b * NP + n) * 3;
  int j = idx33[((long long)b * NP + n) * 33 + k];     // first KK incl self
  const float* nz = nxyz + ((long long)b * NP + j) * 3;
  long long S = (long long)NP * KK;
  long long base = (long long)b * 9 * S + (long long)n * KK + k;
  for (int c = 0; c < 3; ++c) {
    pf[base + (long long)c * S]       = pz[c];
    pf[base + (long long)(3 + c) * S] = nz[c];
    pf[base + (long long)(6 + c) * S] = nz[c] - pz[c];
  }
}

__global__ void f0_build_k(const float* __restrict__ h2, const int* __restrict__ idx33,
                           float* __restrict__ f0)
{
  long long i = (long long)blockIdx.x * blockDim.x + threadIdx.x;
  if (i >= (long long)Bn * Pp * NP * KK) return;
  int k = (int)(i % KK);
  int n = (int)((i / KK) % NP);
  int c = (int)((i / ((long long)KK * NP)) % Pp);
  int b = (int)(i / ((long long)KK * NP * Pp));
  int j = idx33[((long long)b * NP + n) * 33 + k];
  f0[i] = h2[((long long)b * Pp + c) * NP + j] - h2[((long long)b * Pp + c) * NP + n];
}

__global__ void kmean_k(const float* __restrict__ f, float* __restrict__ hl)
{
  int i = blockIdx.x * blockDim.x + threadIdx.x;
  if (i >= Bn * Pp * NP) return;
  const float* p = f + (long long)i * KK;
  float s = 0.f;
  for (int k = 0; k < KK; ++k) s += p[k];
  hl[i] = s * (1.f / KK);
}

// final: lrelu(bn(out_raw) + bn(sc_raw))
__global__ void final_k(const float* __restrict__ o, const float* __restrict__ sc,
                        const float* __restrict__ st1, const float* __restrict__ g1,
                        const float* __restrict__ b1, const float* __restrict__ st2,
                        const float* __restrict__ g2, const float* __restrict__ b2,
                        float* __restrict__ out)
{
  long long i = (long long)blockIdx.x * blockDim.x + threadIdx.x;
  if (i >= (long long)Bn * COUT * NP) return;
  int c = (int)((i / NP) % COUT);
  float v = (o[i]  - st1[2 * c]) * st1[2 * c + 1] * g1[c] + b1[c]
          + (sc[i] - st2[2 * c]) * st2[2 * c + 1] * g2[c] + b2[c];
  out[i] = v > 0.f ? v : 0.2f * v;
}

// =====================================================================
// host orchestration
// =====================================================================
static void gemm(hipStream_t st, const float* A, const float* B, float* C,
                 int M, int N, int K, int lda, int ldb, int ldc,
                 long long sA, long long sB, long long sC, int transA, int batch)
{
  dim3 g((N + BN - 1) / BN, (M + BM - 1) / BM, batch);
  gemm_wmma_k<<<g, dim3(128), 0, st>>>(A, B, C, M, N, K, lda, ldb, ldc, sA, sB, sC, transA);
}

extern "C" void kernel_launch(void* const* d_in, const int* in_sizes, int n_in,
                              void* d_out, int out_size, void* d_ws, size_t ws_size,
                              hipStream_t stream)
{
  (void)in_sizes; (void)n_in; (void)out_size; (void)ws_size;
  const float* xyz      = (const float*)d_in[0];
  const float* x_in     = (const float*)d_in[1];
  const float* conv1_w  = (const float*)d_in[2];
  const float* conv1_g  = (const float*)d_in[3];
  const float* conv1_b  = (const float*)d_in[4];
  const float* conv2_w  = (const float*)d_in[5];
  const float* conv2_g  = (const float*)d_in[6];
  const float* conv2_b  = (const float*)d_in[7];
  const float* sc_w     = (const float*)d_in[8];
  const float* sc_g     = (const float*)d_in[9];
  const float* sc_b     = (const float*)d_in[10];
  const float* cg_att_w = (const float*)d_in[11];
  const float* wk_aw    = (const float*)d_in[12];
  const float* wk_ag    = (const float*)d_in[13];
  const float* wk_ab    = (const float*)d_in[14];
  const float* wk_mw    = (const float*)d_in[15];
  const float* wk_mg    = (const float*)d_in[16];
  const float* wk_mb    = (const float*)d_in[17];
  const float* ca_la    = (const float*)d_in[18];
  const float* ca_a     = (const float*)d_in[19];
  const float* ca_b     = (const float*)d_in[20];
  const float* ca_c     = (const float*)d_in[21];
  const float* ca_n     = (const float*)d_in[22];
  const float* ca_l     = (const float*)d_in[23];
  const float* ca_d     = (const float*)d_in[24];
  const float* ca_dg    = (const float*)d_in[25];
  const float* ca_db    = (const float*)d_in[26];
  const float* lp_xw    = (const float*)d_in[27];
  const float* lp_xg    = (const float*)d_in[28];
  const float* lp_xb    = (const float*)d_in[29];
  const float* lp_m1w   = (const float*)d_in[30];
  const float* lp_m1g   = (const float*)d_in[31];
  const float* lp_m1b   = (const float*)d_in[32];
  const float* lp_m2w   = (const float*)d_in[33];
  const float* lp_m2g   = (const float*)d_in[34];
  const float* lp_m2b   = (const float*)d_in[35];

  float* out_xyzd = (float*)d_out;                        // (B,3,NP)
  float* out_feat = (float*)d_out + (size_t)Bn * 3 * NP;  // (B,COUT,NP)

  // -------- workspace layout (~186 MB) --------
  size_t off = 0;
  auto AF = [&](size_t n) { float* p = (float*)((char*)d_ws + off); off += ((n * 4 + 255) / 256) * 256; return p; };
  auto AI = [&](size_t n) { int*   p = (int*)  ((char*)d_ws + off); off += ((n * 4 + 255) / 256) * 256; return p; };

  int*   fidx   = AI((size_t)Bn * NP);
  int*   gidx   = AI((size_t)Bn * NP * KK);
  int*   idx33  = AI((size_t)Bn * NP * 33);
  int*   start  = AI((size_t)Bn * CN);
  int*   curidx = AI((size_t)Bn * CN);
  float* nxyz   = AF((size_t)Bn * NP * 3);
  float* xnew   = AF((size_t)Bn * CIN * NP);
  float* h      = AF((size_t)Bn * Pp * NP);
  float* xatt   = AF((size_t)Bn * NP);
  float* hx     = AF((size_t)Bn * Pp * NP);
  float* pre    = AF((size_t)Bn * Pp * CN);
  float* cur0   = AF((size_t)Bn * Pp * CN);
  float* cur1   = AF((size_t)Bn * Pp * CN);
  float* mo     = AF((size_t)Bn * 2 * CN);
  float* logits = AF((size_t)Bn * CN * KK);
  float* curves = AF((size_t)Bn * Pp * CN * CL);
  float* att    = AF((size_t)Bn * CN * CL);
  float* sml    = AF((size_t)Bn * CN * CL);
  float* sms    = AF((size_t)Bn * CN * CL);
  float* inter0 = AF((size_t)Bn * Pp * CN);
  float* intra0 = AF((size_t)Bn * Pp * CL);
  float* interM = AF((size_t)Bn * MIDc * CN);
  float* intraM = AF((size_t)Bn * MIDc * CL);
  float* wcx    = AF((size_t)Bn * MIDc * NP);
  float* Wn2    = AF((size_t)Bn * MIDc * CN);
  float* Wl2    = AF((size_t)Bn * MIDc * CL);
  float* XI     = AF((size_t)Bn * CN * NP);
  float* XA     = AF((size_t)Bn * CL * NP);
  float* cfT    = AF((size_t)Bn * Pp * NP);
  float* draw   = AF((size_t)Bn * Pp * NP);
  float* h2     = AF((size_t)Bn * Pp * NP);
  float* pf     = AF((size_t)Bn * 9 * NP * KK);
  float* big1   = AF((size_t)Bn * Pp * NP * KK);
  float* big2   = AF((size_t)Bn * Pp * NP * KK);
  float* hl     = AF((size_t)Bn * Pp * NP);
  float* oraw   = AF((size_t)Bn * COUT * NP);
  float* scraw  = AF((size_t)Bn * COUT * NP);
  float* st_c1  = AF(2 * Pp);
  float* st_mom = AF(4);
  float* st_ag  = AF(2);
  float* st_cad = AF(2 * Pp);
  float* st_x2f = AF(2 * Pp);
  float* st_m1  = AF(2 * Pp);
  float* st_m2  = AF(2 * Pp);
  float* st_o   = AF(2 * COUT);
  float* st_sc  = AF(2 * COUT);

  const long long S_lp = (long long)NP * KK;            // 32768
  auto blks = [](long long n, int t) { return (unsigned)((n + t - 1) / t); };

  // ---- sampling / grouping ----
  fps_k<<<Bn, 1024, 0, stream>>>(xyz, fidx);
  newxyz_k<<<blks(Bn * NP, 256), 256, 0, stream>>>(xyz, fidx, nxyz, out_xyzd);
  ballq_k<<<blks((long long)Bn * NP * 32, 128), 128, 0, stream>>>(xyz, nxyz, gidx);
  groupmax_k<<<Bn * NP, CIN, 0, stream>>>(x_in, gidx, xnew);

  // ---- conv1 + BN + lrelu ----
  gemm(stream, conv1_w, xnew, h, Pp, NP, CIN, CIN, NP, NP, 0, (long long)CIN * NP, (long long)Pp * NP, 0, Bn);
  bn_stats_k<<<Pp, 256, 0, stream>>>(h, st_c1, Bn, Pp, NP);
  bn_apply_k<<<blks((long long)Bn * Pp * NP, 256), 256, 0, stream>>>(
      h, nullptr, h, st_c1, conv1_g, conv1_b, Pp, NP, (long long)Bn * Pp * NP, 1);

  // ---- attention gate / kNN / starts ----
  xatt_k<<<blks(Bn * NP, 256), 256, 0, stream>>>(h, cg_att_w, xatt);
  scale_hx_k<<<blks((long long)Bn * Pp * NP, 256), 256, 0, stream>>>(h, xatt, hx);
  knn_k<<<Bn * NP, 256, 0, stream>>>(nxyz, idx33);
  topk_start_k<<<Bn, 256, 0, stream>>>(xatt, start);

  // ---- curve walk (5 sequential steps) ----
  walk_init_k<<<blks((long long)Bn * Pp * CN, 256), 256, 0, stream>>>(hx, start, pre, curidx);
  for (int step = 0; step < CL; ++step) {
    float* curP = (step & 1) ? cur0 : cur1;   // previous step output
    float* curN = (step & 1) ? cur1 : cur0;   // this step output
    if (step > 0) {
      walk_mom1_k<<<blks(Bn * CN, 256), 256, 0, stream>>>(curP, pre, wk_mw, mo);
      bn_stats_k<<<2, 256, 0, stream>>>(mo, st_mom, Bn, 2, CN);
      walk_mom2_k<<<blks((long long)Bn * Pp * CN, 256), 256, 0, stream>>>(
          mo, st_mom, wk_mg, wk_mb, curP, pre);
    }
    walk_agent1_k<<<Bn * CN, KK, 0, stream>>>(hx, pre, idx33, curidx, wk_aw, logits);
    bn_stats_k<<<1, 256, 0, stream>>>(logits, st_ag, Bn, 1, (long long)CN * KK);
    walk_agent2_k<<<Bn * CN, KK, 0, stream>>>(hx, pre, idx33, curidx, logits, st_ag,
                                              wk_ag, wk_ab, curP, curN, curves, step);
  }

  // ---- curve aggregation ----
  ca_att_k<<<blks(Bn * CN * CL, 256), 256, 0, stream>>>(curves, ca_la, att);
  ca_sml_k<<<blks(Bn * CN, 256), 256, 0, stream>>>(att, sml);
  ca_sms_k<<<1, 256, 0, stream>>>(att, sms);
  ca_inter0_k<<<blks((long long)Bn * Pp * CN, 256), 256, 0, stream>>>(curves, sml, inter0);
  ca_intra0_k<<<blks((long long)Bn * Pp * CL, 256), 256, 0, stream>>>(curves, sms, intra0);

  gemm(stream, ca_a, inter0, interM, MIDc, CN, Pp, Pp, CN, CN, 0, (long long)Pp * CN, (long long)MIDc * CN, 0, Bn);
  gemm(stream, ca_b, intra0, intraM, MIDc, CL, Pp, Pp, CL, CL, 0, (long long)Pp * CL, (long long)MIDc * CL, 0, Bn);
  gemm(stream, ca_c, h, wcx, MIDc, NP, Pp, Pp, NP, NP, 0, (long long)Pp * NP, (long long)MIDc * NP, 0, Bn);
  gemm(stream, ca_n, interM, Wn2, MIDc, CN, MIDc, MIDc, CN, CN, 0, (long long)MIDc * CN, (long long)MIDc * CN, 0, Bn);
  gemm(stream, ca_l, intraM, Wl2, MIDc, CL, MIDc, MIDc, CL, CL, 0, (long long)MIDc * CL, (long long)MIDc * CL, 0, Bn);
  // XI[q,n] = sum_m interM[m,q] * wcx[m,n]  (A transposed)
  gemm(stream, interM, wcx, XI, CN, NP, MIDc, CN, NP, NP,
       (long long)MIDc * CN, (long long)MIDc * NP, (long long)CN * NP, 1, Bn);
  gemm(stream, intraM, wcx, XA, CL, NP, MIDc, CL, NP, NP,
       (long long)MIDc * CL, (long long)MIDc * NP, (long long)CL * NP, 1, Bn);
  softmax_col_k<<<blks(Bn * NP, 256), 256, 0, stream>>>(XI, CN, NP, (long long)CN * NP, NP, Bn);
  softmax_col_k<<<blks(Bn * NP, 256), 256, 0, stream>>>(XA, CL, NP, (long long)CL * NP, NP, Bn);
  // cfT rows 0..31 and 32..63
  gemm(stream, Wn2, XI, cfT, MIDc, NP, CN, CN, NP, NP,
       (long long)MIDc * CN, (long long)CN * NP, (long long)Pp * NP, 0, Bn);
  gemm(stream, Wl2, XA, cfT + (size_t)MIDc * NP, MIDc, NP, CL, CL, NP, NP,
       (long long)MIDc * CL, (long long)CL * NP, (long long)Pp * NP, 0, Bn);
  gemm(stream, ca_d, cfT, draw, Pp, NP, Pp, Pp, NP, NP, 0, (long long)Pp * NP, (long long)Pp * NP, 0, Bn);
  bn_stats_k<<<Pp, 256, 0, stream>>>(draw, st_cad, Bn, Pp, NP);
  bn_apply_k<<<blks((long long)Bn * Pp * NP, 256), 256, 0, stream>>>(
      draw, h, h2, st_cad, ca_dg, ca_db, Pp, NP, (long long)Bn * Pp * NP, 1);

  // ---- LPFA ----
  pf_build_k<<<blks((long long)Bn * NP * KK, 256), 256, 0, stream>>>(nxyz, idx33, pf);
  f0_build_k<<<blks((long long)Bn * Pp * S_lp, 256), 256, 0, stream>>>(h2, idx33, big1);
  gemm(stream, lp_xw, pf, big2, Pp, (int)S_lp, 9, 9, (int)S_lp, (int)S_lp,
       0, 9 * S_lp, (long long)Pp * S_lp, 0, Bn);
  bn_stats_k<<<Pp, 256, 0, stream>>>(big2, st_x2f, Bn, Pp, S_lp);
  bn_apply_k<<<blks((long long)Bn * Pp * S_lp, 256), 256, 0, stream>>>(
      big2, big1, big1, st_x2f, lp_xg, lp_xb, Pp, S_lp, (long long)Bn * Pp * S_lp, 1);
  gemm(stream, lp_m1w, big1, big2, Pp, (int)S_lp, Pp, Pp, (int)S_lp, (int)S_lp,
       0, (long long)Pp * S_lp, (long long)Pp * S_lp, 0, Bn);
  bn_stats_k<<<Pp, 256, 0, stream>>>(big2, st_m1, Bn, Pp, S_lp);
  bn_apply_k<<<blks((long long)Bn * Pp * S_lp, 256), 256, 0, stream>>>(
      big2, nullptr, big1, st_m1, lp_m1g, lp_m1b, Pp, S_lp, (long long)Bn * Pp * S_lp, 1);
  gemm(stream, lp_m2w, big1, big2, Pp, (int)S_lp, Pp, Pp, (int)S_lp, (int)S_lp,
       0, (long long)Pp * S_lp, (long long)Pp * S_lp, 0, Bn);
  bn_stats_k<<<Pp, 256, 0, stream>>>(big2, st_m2, Bn, Pp, S_lp);
  bn_apply_k<<<blks((long long)Bn * Pp * S_lp, 256), 256, 0, stream>>>(
      big2, nullptr, big2, st_m2, lp_m2g, lp_m2b, Pp, S_lp, (long long)Bn * Pp * S_lp, 1);
  kmean_k<<<blks((long long)Bn * Pp * NP, 256), 256, 0, stream>>>(big2, hl);

  // ---- conv2 + shortcut + fuse ----
  gemm(stream, conv2_w, hl, oraw, COUT, NP, Pp, Pp, NP, NP, 0, (long long)Pp * NP, (long long)COUT * NP, 0, Bn);
  gemm(stream, sc_w, xnew, scraw, COUT, NP, CIN, CIN, NP, NP, 0, (long long)CIN * NP, (long long)COUT * NP, 0, Bn);
  bn_stats_k<<<COUT, 256, 0, stream>>>(oraw, st_o, Bn, COUT, NP);
  bn_stats_k<<<COUT, 256, 0, stream>>>(scraw, st_sc, Bn, COUT, NP);
  final_k<<<blks((long long)Bn * COUT * NP, 256), 256, 0, stream>>>(
      oraw, scraw, st_o, conv2_g, conv2_b, st_sc, sc_g, sc_b, out_feat);
}
